// MwAN_snli_58394375357398
// MI455X (gfx1250) — matrix operand order repaired and checked by reading, not compiled, not run on hardware
//
#include <hip/hip_runtime.h>
#include <math.h>

typedef __attribute__((ext_vector_type(16))) _Float16 v16h;
typedef __attribute__((ext_vector_type(8)))  _Float16 v8h;
typedef __attribute__((ext_vector_type(4)))  _Float16 v4h;
typedef __attribute__((ext_vector_type(8)))  float    v8f;

#define BATCH 16
#define SEQ   128
#define EMBD  300
#define HID   150
#define H2    300
#define H3    450
#define H12   1800
#define ROWS  (BATCH*SEQ)   // 2048

// concat two v8h into a v16h fragment
__device__ __forceinline__ v16h frag_cat(v8h lo, v8h hi) {
  return __builtin_shufflevector(lo, hi, 0,1,2,3,4,5,6,7,8,9,10,11,12,13,14,15);
}

__device__ __forceinline__ v4h cvt4(float4 f) {
  v4h h;
  h[0] = (_Float16)f.x; h[1] = (_Float16)f.y;
  h[2] = (_Float16)f.z; h[3] = (_Float16)f.w;
  return h;
}

// ---------------------------------------------------------------------------
// Embedding gather (float4 vectorized): xp = emb[premise], xh = emb[hypothesis]
// ---------------------------------------------------------------------------
__global__ __launch_bounds__(256) void mwan_embed(const int* __restrict__ prem,
                                                  const int* __restrict__ hyp,
                                                  const float* __restrict__ emb,
                                                  float* __restrict__ xp,
                                                  float* __restrict__ xh) {
  size_t c = (size_t)blockIdx.x * blockDim.x + threadIdx.x;   // float4 chunk id
  const size_t rowchunks = EMBD / 4;                          // 75
  const size_t total = (size_t)ROWS * rowchunks;
  if (c >= 2 * total) return;
  int which = c >= total;
  size_t r = which ? (c - total) : c;
  int row = (int)(r / rowchunks), e4 = (int)(r % rowchunks) * 4;
  int tok = which ? hyp[row] : prem[row];
  float4 v = *(const float4*)(emb + (size_t)tok * EMBD + e4);
  *(float4*)((which ? xh : xp) + (size_t)row * EMBD + e4) = v;
}

// ---------------------------------------------------------------------------
// Generic WMMA GEMM:  C[M,N] = A[M,K] @ W[N,K]^T + bias[N]
// Requirements: M % 128 == 0, K % 4 == 0, A/W rows 16B aligned (K*4 % 16 == 0).
// Block: 256 threads (8 waves). Tile 128(M) x 32(N), K step 32.
// ---------------------------------------------------------------------------
__global__ __launch_bounds__(256) void mwan_gemm(const float* __restrict__ A,
                                                 const float* __restrict__ W,
                                                 const float* __restrict__ bias,
                                                 float* __restrict__ C,
                                                 int M, int N, int K) {
  __shared__ __align__(16) _Float16 As[128][40];    // A[m][k], 16B-aligned rows
  __shared__ __align__(16) _Float16 BsT[32][40];    // B^T: [n][k]
  int tid = threadIdx.x;
  int wave = tid >> 5, lane = tid & 31;
  int g = lane >> 4, lr = lane & 15;
  int m0 = blockIdx.x * 128;
  int n0 = blockIdx.y * 32;
  v8f acc0 = {}, acc1 = {};
  int ktiles = (K + 31) >> 5;
  for (int kt = 0; kt < ktiles; ++kt) {
    int k0 = kt << 5;
    __syncthreads();
    // prefetch next K tile of A (speculative; dropped if OOB)
    if (kt + 1 < ktiles) {
      int nk = k0 + 32 + (tid & 1) * 16;
      if (nk < K) __builtin_prefetch(A + (size_t)(m0 + (tid >> 1)) * K + nk, 0, 1);
    }
    // stage A: 128x32 in float4 chunks (4 chunks / thread)
#pragma unroll
    for (int it = 0; it < 4; ++it) {
      int c = tid + 256 * it;
      int row = c >> 3, col4 = (c & 7) << 2;
      int gk = k0 + col4;
      v4h h4 = {};
      if (gk < K)   // chunk fully valid (K % 4 == 0)
        h4 = cvt4(*(const float4*)(A + (size_t)(m0 + row) * K + gk));
      *(v4h*)&As[row][col4] = h4;
    }
    // stage B^T: BsT[n][k] = W[n0+n][k0+k], 1 chunk / thread
    {
      int nn = tid >> 3, col4 = (tid & 7) << 2;
      int gn = n0 + nn, gk = k0 + col4;
      v4h h4 = {};
      if (gn < N && gk < K)
        h4 = cvt4(*(const float4*)(W + (size_t)gn * K + gk));
      *(v4h*)&BsT[nn][col4] = h4;
    }
    __syncthreads();
    // fragments: contiguous 16B chunks -> ds_load_b128
    const _Float16* arow = &As[wave * 16 + lr][0];
    v16h af = frag_cat(*(const v8h*)(arow + 16 * g), *(const v8h*)(arow + 16 * g + 8));
    const _Float16* b0 = &BsT[lr][0];
    const _Float16* b1 = &BsT[16 + lr][0];
    v16h bf0 = frag_cat(*(const v8h*)(b0 + 16 * g), *(const v8h*)(b0 + 16 * g + 8));
    v16h bf1 = frag_cat(*(const v8h*)(b1 + 16 * g), *(const v8h*)(b1 + 16 * g + 8));
    acc0 = __builtin_amdgcn_wmma_f32_16x16x32_f16(false, af, false, bf0, (short)0, acc0, false, false);
    acc1 = __builtin_amdgcn_wmma_f32_16x16x32_f16(false, af, false, bf1, (short)0, acc1, false, false);
  }
  // epilogue: C layout VGPR r -> M = r + 8*(lane>=16), N = lane%16
#pragma unroll
  for (int r = 0; r < 8; ++r) {
    int gm = m0 + wave * 16 + r + 8 * g;
    if (gm >= M) continue;
    int gn0 = n0 + lr, gn1 = n0 + 16 + lr;
    if (gn0 < N) C[(size_t)gm * N + gn0] = acc0[r] + (bias ? bias[gn0] : 0.f);
    if (gn1 < N) C[(size_t)gm * N + gn1] = acc1[r] + (bias ? bias[gn1] : 0.f);
  }
}

// ---------------------------------------------------------------------------
// Shared helpers: softmax over 128 values in LDS; weighted sum over rows
// ---------------------------------------------------------------------------
__device__ __forceinline__ void softmax128(float* sj, float* red, int tid) {
  red[tid] = (tid < 128) ? sj[tid] : -3.4e38f;
  __syncthreads();
  for (int s = 128; s >= 1; s >>= 1) {
    if (tid < s) red[tid] = fmaxf(red[tid], red[tid + s]);
    __syncthreads();
  }
  float mx = red[0];
  __syncthreads();
  float e = 0.f;
  if (tid < 128) { e = expf(sj[tid] - mx); sj[tid] = e; }
  red[tid] = (tid < 128) ? e : 0.f;
  __syncthreads();
  for (int s = 128; s >= 1; s >>= 1) {
    if (tid < s) red[tid] += red[tid + s];
    __syncthreads();
  }
  float inv = 1.f / red[0];
  __syncthreads();
  if (tid < 128) sj[tid] *= inv;
  __syncthreads();
}

__device__ __forceinline__ void wsum_rows(const float* sj, const float* __restrict__ X,
                                          int b, float* __restrict__ outrow, int tid) {
  for (int d = tid; d < H2; d += 256) {
    float s = 0.f;
    for (int j = 0; j < SEQ; ++j) s += sj[j] * X[((size_t)b * SEQ + j) * H2 + d];
    outrow[d] = s;
  }
}

// ---------------------------------------------------------------------------
// Fused WMMA attention for dot/minus:  per block (b,t):
//   A[j,k] = f(X[b,j,k], Y[b,t,k]) ; S = A @ W^T ; s_j = sum_n v[n]*tanh(S[j,n])
//   p = softmax_j(s) ; Out[b,t,:] = sum_j p_j X[b,j,:]
// mode 0: multiply, mode 1: subtract
// ---------------------------------------------------------------------------
__global__ __launch_bounds__(256) void mwan_attn_wmma(const float* __restrict__ X,
                                                      const float* __restrict__ Y,
                                                      const float* __restrict__ Wt,   // (150,300)
                                                      const float* __restrict__ vvec, // (150)
                                                      float* __restrict__ Out,
                                                      int mode) {
  __shared__ __align__(16) float    yt[H2];
  __shared__ __align__(16) _Float16 As[128][40];
  __shared__ __align__(16) _Float16 BsT[160][40];   // W^T tile: [n][k]
  __shared__ float vl[160];
  __shared__ float sj[128];
  __shared__ float red[256];
  int b = blockIdx.x >> 7, t = blockIdx.x & 127;
  int tid = threadIdx.x;
  int wave = tid >> 5, lane = tid & 31;
  int g = lane >> 4, lr = lane & 15;

  const float* yrow = Y + ((size_t)b * SEQ + t) * H2;
  for (int i = tid; i < H2; i += 256) yt[i] = yrow[i];
  for (int i = tid; i < 160; i += 256) vl[i] = (i < HID) ? vvec[i] : 0.f;
  if (tid < 128) sj[tid] = 0.f;

  v8f acc[10] = {};
  for (int kt = 0; kt < 10; ++kt) {          // K = 300 -> 10 tiles of 32 (padded)
    int k0 = kt << 5;
    __syncthreads();
    if (kt < 9) {
      int nk = k0 + 32 + (tid & 1) * 16;
      if (nk < H2) __builtin_prefetch(X + ((size_t)b * SEQ + (tid >> 1)) * H2 + nk, 0, 1);
    }
    // stage A = f(X, y_t): 128x32, float4 chunks (chunks fully valid: 300 % 4 == 0)
#pragma unroll
    for (int it = 0; it < 4; ++it) {
      int c = tid + 256 * it;
      int row = c >> 3, col4 = (c & 7) << 2;
      int gk = k0 + col4;
      v4h h4 = {};
      if (gk < H2) {
        float4 x4 = *(const float4*)(X + ((size_t)b * SEQ + row) * H2 + gk);
        float4 y4 = *(const float4*)(yt + gk);
        if (mode) { x4.x -= y4.x; x4.y -= y4.y; x4.z -= y4.z; x4.w -= y4.w; }
        else      { x4.x *= y4.x; x4.y *= y4.y; x4.z *= y4.z; x4.w *= y4.w; }
        h4 = cvt4(x4);
      }
      *(v4h*)&As[row][col4] = h4;
    }
    // stage B^T: BsT[n][k] = Wt[n][k0+k], 160x32, 5 chunks / thread
#pragma unroll
    for (int it = 0; it < 5; ++it) {
      int c = tid + 256 * it;
      int nn = c >> 3, col4 = (c & 7) << 2;
      int gk = k0 + col4;
      v4h h4 = {};
      if (nn < HID && gk < H2)
        h4 = cvt4(*(const float4*)(Wt + (size_t)nn * H2 + gk));
      *(v4h*)&BsT[nn][col4] = h4;
    }
    __syncthreads();
    const _Float16* arow = &As[wave * 16 + lr][0];
    v16h af = frag_cat(*(const v8h*)(arow + 16 * g), *(const v8h*)(arow + 16 * g + 8));
#pragma unroll
    for (int nt = 0; nt < 10; ++nt) {
      const _Float16* brow = &BsT[nt * 16 + lr][0];
      v16h bf = frag_cat(*(const v8h*)(brow + 16 * g), *(const v8h*)(brow + 16 * g + 8));
      acc[nt] = __builtin_amdgcn_wmma_f32_16x16x32_f16(false, af, false, bf, (short)0, acc[nt], false, false);
    }
  }
  __syncthreads();
  // s_j = sum_n v[n]*tanh(S[j,n]); lane (g,lr) owns n = 16*nt + lr, row j = 16*wave + r + 8*g
#pragma unroll
  for (int r = 0; r < 8; ++r) {
    int j = wave * 16 + r + 8 * g;
    float part = 0.f;
#pragma unroll
    for (int nt = 0; nt < 10; ++nt) part += vl[nt * 16 + lr] * tanhf(acc[nt][r]);
    atomicAdd(&sj[j], part);
  }
  __syncthreads();
  softmax128(sj, red, tid);
  wsum_rows(sj, X, b, Out + ((size_t)b * SEQ + t) * H2, tid);
}

// ---------------------------------------------------------------------------
// Concat attention: s_j = sum_n vc[n]*tanh(Pc[b,j,n] + Hc[b,t,n]); out = softmax @ hp
// ---------------------------------------------------------------------------
__global__ __launch_bounds__(256) void mwan_attn_concat(const float* __restrict__ Pc,
                                                        const float* __restrict__ Hc,
                                                        const float* __restrict__ vc,
                                                        const float* __restrict__ hp,
                                                        float* __restrict__ Out) {
  __shared__ __align__(16) float ht[HID];
  __shared__ float sj[128];
  __shared__ float red[256];
  int b = blockIdx.x >> 7, t = blockIdx.x & 127, tid = threadIdx.x;
  for (int i = tid; i < HID; i += 256) ht[i] = Hc[((size_t)b * SEQ + t) * HID + i];
  __syncthreads();
  if (tid < 128) {
    const float* pr = Pc + ((size_t)b * SEQ + tid) * HID;
    float s = 0.f;
    for (int n = 0; n < HID; ++n) s += vc[n] * tanhf(pr[n] + ht[n]);
    sj[tid] = s;
  }
  __syncthreads();
  softmax128(sj, red, tid);
  wsum_rows(sj, hp, b, Out + ((size_t)b * SEQ + t) * H2, tid);
}

// ---------------------------------------------------------------------------
// Bilinear attention: s_j = hh[b,t,:] . Pb[b,j,:]; out = softmax @ hp
// ---------------------------------------------------------------------------
__global__ __launch_bounds__(256) void mwan_attn_bilin(const float* __restrict__ hh,
                                                       const float* __restrict__ Pb,
                                                       const float* __restrict__ hp,
                                                       float* __restrict__ Out) {
  __shared__ __align__(16) float yt[H2];
  __shared__ float sj[128];
  __shared__ float red[256];
  int b = blockIdx.x >> 7, t = blockIdx.x & 127, tid = threadIdx.x;
  for (int i = tid; i < H2; i += 256) yt[i] = hh[((size_t)b * SEQ + t) * H2 + i];
  __syncthreads();
  if (tid < 128) {
    const float* pr = Pb + ((size_t)b * SEQ + tid) * H2;
    float s = 0.f;
    for (int d = 0; d < H2; ++d) s += yt[d] * pr[d];
    sj[tid] = s;
  }
  __syncthreads();
  softmax128(sj, red, tid);
  wsum_rows(sj, hp, b, Out + ((size_t)b * SEQ + t) * H2, tid);
}

// ---------------------------------------------------------------------------
// Sequential GRU (one direction). grid = B, block = 256. gi precomputed.
// hout row stride 2H, column offset coloff (0 = fwd half, H = bwd half).
// ---------------------------------------------------------------------------
__global__ __launch_bounds__(256) void mwan_gru(const float* __restrict__ gi,
                                                const float* __restrict__ Whh,  // (450,150)
                                                const float* __restrict__ bhh,  // (450)
                                                float* __restrict__ hout,
                                                int reverse, int coloff) {
  __shared__ __align__(16) float h[HID];
  __shared__ float gh[H3];
  int b = blockIdx.x, tid = threadIdx.x;
  for (int i = tid; i < HID; i += 256) h[i] = 0.f;
  __syncthreads();
  for (int s = 0; s < SEQ; ++s) {
    int t = reverse ? (SEQ - 1 - s) : s;
    for (int n = tid; n < H3; n += 256) {
      float acc = bhh[n];
      const float2* wr2 = (const float2*)(Whh + (size_t)n * HID);   // 600n bytes: 8B aligned
      const float2* h2 = (const float2*)h;
      for (int d = 0; d < HID / 2; ++d) {
        float2 w = wr2[d], hv = h2[d];
        acc += w.x * hv.x + w.y * hv.y;
      }
      gh[n] = acc;
    }
    __syncthreads();
    if (tid < HID) {
      const float* gir = gi + ((size_t)b * SEQ + t) * H3;
      float ir = gir[tid], iz = gir[HID + tid], inn = gir[2 * HID + tid];
      float r = 1.f / (1.f + expf(-(ir + gh[tid])));
      float z = 1.f / (1.f + expf(-(iz + gh[HID + tid])));
      float nn = tanhf(inn + r * gh[2 * HID + tid]);
      float hn = (1.f - z) * nn + z * h[tid];
      h[tid] = hn;
      hout[((size_t)b * SEQ + t) * H2 + coloff + tid] = hn;
    }
    __syncthreads();
  }
}

// ---------------------------------------------------------------------------
// Concat agg = [hh | qts | qtc | qtd | qtb | qtm]  (B*L, 1800), float4 copies
// ---------------------------------------------------------------------------
__global__ __launch_bounds__(256) void mwan_concat(const float* __restrict__ hh,
                                                   const float* __restrict__ qts,
                                                   const float* __restrict__ qtc,
                                                   const float* __restrict__ qtd,
                                                   const float* __restrict__ qtb,
                                                   const float* __restrict__ qtm,
                                                   float* __restrict__ agg) {
  size_t c = (size_t)blockIdx.x * blockDim.x + threadIdx.x;    // float4 chunk
  const int rowchunks = H12 / 4;                               // 450
  if (c >= (size_t)ROWS * rowchunks) return;
  int row = (int)(c / rowchunks), q = (int)(c % rowchunks);
  int seg = q / (H2 / 4), off4 = (q % (H2 / 4)) * 4;
  const float* src;
  switch (seg) {
    case 0: src = hh;  break;
    case 1: src = qts; break;
    case 2: src = qtc; break;
    case 3: src = qtd; break;
    case 4: src = qtb; break;
    default: src = qtm; break;
  }
  float4 v = *(const float4*)(src + (size_t)row * H2 + off4);
  *(float4*)(agg + (size_t)row * H12 + seg * H2 + off4) = v;
}

// ---------------------------------------------------------------------------
// Final head: rq pooling over hp, rp pooling over aggr, prediction + leaky relu
// ---------------------------------------------------------------------------
__global__ __launch_bounds__(256) void mwan_final(const float* __restrict__ Q,     // (B*L,150) = hp@Wq^T
                                                  const float* __restrict__ hp,
                                                  const float* __restrict__ vq,
                                                  const float* __restrict__ aggr,
                                                  const float* __restrict__ P1,    // (B*L,150) = aggr@Wp1^T
                                                  const float* __restrict__ Wp2,   // (150,300)
                                                  const float* __restrict__ vp,
                                                  const float* __restrict__ Wpred, // (3,300)
                                                  float* __restrict__ out) {
  __shared__ float st[128];
  __shared__ float red[256];
  __shared__ float rq[H2];
  __shared__ float w2[HID];
  __shared__ float rp[H2];
  int b = blockIdx.x, tid = threadIdx.x;
  if (tid < 128) {
    const float* qr = Q + ((size_t)b * SEQ + tid) * HID;
    float s = 0.f;
    for (int n = 0; n < HID; ++n) s += vq[n] * tanhf(qr[n]);
    st[tid] = s;
  }
  __syncthreads();
  softmax128(st, red, tid);
  for (int d = tid; d < H2; d += 256) {
    float s = 0.f;
    for (int t = 0; t < SEQ; ++t) s += st[t] * hp[((size_t)b * SEQ + t) * H2 + d];
    rq[d] = s;
  }
  __syncthreads();
  if (tid < HID) {
    const float* wr = Wp2 + (size_t)tid * H2;
    float s = 0.f;
    for (int d = 0; d < H2; ++d) s += wr[d] * rq[d];
    w2[tid] = s;
  }
  __syncthreads();
  if (tid < 128) {
    const float* pr = P1 + ((size_t)b * SEQ + tid) * HID;
    float s = 0.f;
    for (int n = 0; n < HID; ++n) s += vp[n] * (pr[n] + w2[n]);
    st[tid] = s;
  }
  __syncthreads();
  softmax128(st, red, tid);
  for (int d = tid; d < H2; d += 256) {
    float s = 0.f;
    for (int t = 0; t < SEQ; ++t) s += st[t] * aggr[((size_t)b * SEQ + t) * H2 + d];
    rp[d] = s;
  }
  __syncthreads();
  if (tid < 3) {
    const float* wr = Wpred + (size_t)tid * H2;
    float s = 0.f;
    for (int d = 0; d < H2; ++d) s += wr[d] * rp[d];
    out[b * 3 + tid] = (s > 0.f) ? s : 0.01f * s;   // leaky_relu, slope 0.01
  }
}

// ---------------------------------------------------------------------------
extern "C" void kernel_launch(void* const* d_in, const int* in_sizes, int n_in,
                              void* d_out, int out_size, void* d_ws, size_t ws_size,
                              hipStream_t stream) {
  (void)in_sizes; (void)n_in; (void)out_size; (void)ws_size;
  const int*   premise    = (const int*)d_in[0];
  const int*   hypothesis = (const int*)d_in[1];
  const float* emb        = (const float*)d_in[2];
  const float *pg_Wih_f = (const float*)d_in[3],  *pg_Whh_f = (const float*)d_in[4],
              *pg_bih_f = (const float*)d_in[5],  *pg_bhh_f = (const float*)d_in[6];
  const float *pg_Wih_b = (const float*)d_in[7],  *pg_Whh_b = (const float*)d_in[8],
              *pg_bih_b = (const float*)d_in[9],  *pg_bhh_b = (const float*)d_in[10];
  const float *hg_Wih_f = (const float*)d_in[11], *hg_Whh_f = (const float*)d_in[12],
              *hg_bih_f = (const float*)d_in[13], *hg_bhh_f = (const float*)d_in[14];
  const float *hg_Wih_b = (const float*)d_in[15], *hg_Whh_b = (const float*)d_in[16],
              *hg_bih_b = (const float*)d_in[17], *hg_bhh_b = (const float*)d_in[18];
  const float *ag_Wih_f = (const float*)d_in[19], *ag_Whh_f = (const float*)d_in[20],
              *ag_bih_f = (const float*)d_in[21], *ag_bhh_f = (const float*)d_in[22];
  const float *ag_Wih_b = (const float*)d_in[23], *ag_Whh_b = (const float*)d_in[24],
              *ag_bih_b = (const float*)d_in[25], *ag_bhh_b = (const float*)d_in[26];
  const float *Wc1 = (const float*)d_in[27], *Wc2 = (const float*)d_in[28], *vc = (const float*)d_in[29];
  const float *Wb  = (const float*)d_in[30];
  const float *Wd1 = (const float*)d_in[31], *vd1 = (const float*)d_in[32];
  const float *Wd2 = (const float*)d_in[33], *vd2 = (const float*)d_in[34];
  const float *Wm  = (const float*)d_in[35], *vm  = (const float*)d_in[36];
  const float *Wq  = (const float*)d_in[37], *vq  = (const float*)d_in[38];
  const float *Wp1 = (const float*)d_in[39], *Wp2 = (const float*)d_in[40], *vp = (const float*)d_in[41];
  const float *Wpred = (const float*)d_in[42];
  float* out = (float*)d_out;

  // workspace allocator (fp32 elements, 64-elem aligned)
  float* ws = (float*)d_ws;
  size_t off = 0;
  auto alloc = [&](size_t n) { float* p = ws + off; off += (n + 63) & ~(size_t)63; return p; };
  float* xp  = alloc((size_t)ROWS * EMBD);
  float* xh  = alloc((size_t)ROWS * EMBD);
  float* giA = alloc((size_t)ROWS * H3);
  float* giB = alloc((size_t)ROWS * H3);
  float* hp  = alloc((size_t)ROWS * H2);
  float* hh  = alloc((size_t)ROWS * H2);
  float* Pc  = alloc((size_t)ROWS * HID);
  float* Hc  = alloc((size_t)ROWS * HID);
  float* Pb  = alloc((size_t)ROWS * H2);
  float* qts = alloc((size_t)ROWS * H2);
  float* qtc = alloc((size_t)ROWS * H2);
  float* qtd = alloc((size_t)ROWS * H2);
  float* qtb = alloc((size_t)ROWS * H2);
  float* qtm = alloc((size_t)ROWS * H2);
  float* agg = alloc((size_t)ROWS * H12);
  float* aggr= alloc((size_t)ROWS * H2);
  float* Q   = alloc((size_t)ROWS * HID);
  float* P1  = alloc((size_t)ROWS * HID);

  auto gemm = [&](const float* A, const float* W, const float* bias, float* C,
                  int M, int N, int K) {
    dim3 grid((M + 127) / 128, (N + 31) / 32);
    mwan_gemm<<<grid, 256, 0, stream>>>(A, W, bias, C, M, N, K);
  };

  // 1) embeddings
  {
    size_t total = 2 * (size_t)ROWS * (EMBD / 4);
    mwan_embed<<<(unsigned)((total + 255) / 256), 256, 0, stream>>>(premise, hypothesis, emb, xp, xh);
  }

  // 2) premise BiGRU
  gemm(xp, pg_Wih_f, pg_bih_f, giA, ROWS, H3, EMBD);
  gemm(xp, pg_Wih_b, pg_bih_b, giB, ROWS, H3, EMBD);
  mwan_gru<<<BATCH, 256, 0, stream>>>(giA, pg_Whh_f, pg_bhh_f, hp, 0, 0);
  mwan_gru<<<BATCH, 256, 0, stream>>>(giB, pg_Whh_b, pg_bhh_b, hp, 1, HID);

  // 3) hypothesis BiGRU
  gemm(xh, hg_Wih_f, hg_bih_f, giA, ROWS, H3, EMBD);
  gemm(xh, hg_Wih_b, hg_bih_b, giB, ROWS, H3, EMBD);
  mwan_gru<<<BATCH, 256, 0, stream>>>(giA, hg_Whh_f, hg_bhh_f, hh, 0, 0);
  mwan_gru<<<BATCH, 256, 0, stream>>>(giB, hg_Whh_b, hg_bhh_b, hh, 1, HID);

  // 4) attention projections
  gemm(hp, Wc1, nullptr, Pc, ROWS, HID, H2);
  gemm(hh, Wc2, nullptr, Hc, ROWS, HID, H2);
  gemm(hp, Wb,  nullptr, Pb, ROWS, H2,  H2);

  // 5) attentions (grid = B*L)
  mwan_attn_concat<<<ROWS, 256, 0, stream>>>(Pc, Hc, vc, hp, qtc);
  mwan_attn_bilin <<<ROWS, 256, 0, stream>>>(hh, Pb, hp, qtb);
  mwan_attn_wmma  <<<ROWS, 256, 0, stream>>>(hp, hh, Wd1, vd1, qts, 0); // dot 1
  mwan_attn_wmma  <<<ROWS, 256, 0, stream>>>(hh, hp, Wd2, vd2, qtd, 0); // dot 2
  mwan_attn_wmma  <<<ROWS, 256, 0, stream>>>(hp, hh, Wm,  vm,  qtm, 1); // minus

  // 6) aggregation BiGRU over concat
  {
    size_t total = (size_t)ROWS * (H12 / 4);
    mwan_concat<<<(unsigned)((total + 255) / 256), 256, 0, stream>>>(hh, qts, qtc, qtd, qtb, qtm, agg);
  }
  gemm(agg, ag_Wih_f, ag_bih_f, giA, ROWS, H3, H12);
  gemm(agg, ag_Wih_b, ag_bih_b, giB, ROWS, H3, H12);
  mwan_gru<<<BATCH, 256, 0, stream>>>(giA, ag_Whh_f, ag_bhh_f, aggr, 0, 0);
  mwan_gru<<<BATCH, 256, 0, stream>>>(giB, ag_Whh_b, ag_bhh_b, aggr, 1, HID);

  // 7) final head
  gemm(hp,   Wq,  nullptr, Q,  ROWS, HID, H2);
  gemm(aggr, Wp1, nullptr, P1, ROWS, HID, H2);
  mwan_final<<<BATCH, 256, 0, stream>>>(Q, hp, vq, aggr, P1, Wp2, vp, Wpred, out);
}
